// SetAbstraction2D_74921409511649
// MI455X (gfx1250) — matrix operand order repaired and checked by reading, not compiled, not run on hardware
//
#include <hip/hip_runtime.h>

// ---------------------------------------------------------------------------
// PointNet++ Set Abstraction for MI455X (gfx1250, wave32, WMMA)
//   B=16, N=4096, C=64, NPOINT=1024, K=32, dims 66->64->64->128
// Pipeline: [fold BN into swizzled f16 weights] -> [FPS] -> [wave/center KNN]
//           -> [fused gather + 3x WMMA-GEMM MLP + maxpool]
// Weights are stored fragment-major so each lane's 16 B-halves are one
// contiguous 32B load (2x global_load_b128) instead of 16 scalar loads.
// ---------------------------------------------------------------------------

typedef _Float16 v16h __attribute__((ext_vector_type(16)));
typedef _Float16 v8h  __attribute__((ext_vector_type(8)));
typedef _Float16 h2   __attribute__((ext_vector_type(2)));
typedef float    v8f  __attribute__((ext_vector_type(8)));

#define B_      16
#define N_      4096
#define C_      64
#define S_      1024   // NPOINT
#define K_      32     // neighbors
#define EPSV    1e-5f

// workspace byte offsets
#define OFF_GIDX 0                          // int[B_*S_*K_] = 2 MiB
#define OFF_W1F  (B_*S_*K_*4)               // f16[4*3*512]  (96x64 swizzled)
#define OFF_W2F  (OFF_W1F + 96*64*2)        // f16[4*2*512]  (64x64 swizzled)
#define OFF_W3F  (OFF_W2F + 64*64*2)        // f16[8*2*512]  (64x128 swizzled)
#define OFF_B1F  (OFF_W3F + 64*128*2)       // f32[64]
#define OFF_B2F  (OFF_B1F + 256)            // f32[64]
#define OFF_B3F  (OFF_B2F + 256)            // f32[128]

// d_out float offsets: (center_coords, new_features, center_valid)
#define OUT_CC 0
#define OUT_NF (B_*S_*2)
#define OUT_CV (OUT_NF + B_*S_*128)

// ---- wave32 primitives (avoid HIP wrappers: __AMDGCN_WAVEFRONT_SIZE absent)
__device__ __forceinline__ int bperm_i(int srcLane, int v) {
  return __builtin_amdgcn_ds_bpermute(srcLane << 2, v);
}
__device__ __forceinline__ float bperm_f(int srcLane, float v) {
  return __builtin_bit_cast(float, bperm_i(srcLane, __builtin_bit_cast(int, v)));
}
__device__ __forceinline__ float shflx_f(float v, int m, int lane) { return bperm_f(lane ^ m, v); }
__device__ __forceinline__ int   shflx_i(int   v, int m, int lane) { return bperm_i(lane ^ m, v); }

// ---------------------------------------------------------------------------
// Kernel 0: fold batchnorm into weights; emit fragment-major ("swizzled") f16.
// Fragment f = nt*KS + ks; lane L holds, contiguously, the 16 halves
//   W[k][n]*scale[n], k = ks*32 + ((L&16)?16:0) + i, n = nt*16 + (L&15)
// ---------------------------------------------------------------------------
__device__ __forceinline__ void swizzle_weights(
    _Float16* dst, const float* W, int N, int Kdim, int KS, int NT,
    const float* g, const float* rv, int tid)
{
  int total = NT * KS * 512;                 // 32 lanes * 16 halves per frag
  for (int t = tid; t < total; t += 256) {
    int i  = t & 15;
    int L  = (t >> 4) & 31;
    int f  = t >> 9;
    int ks = f % KS, nt = f / KS;
    int n  = nt * 16 + (L & 15);
    int k  = ks * 32 + ((L & 16) ? 16 : 0) + i;
    float s = g[n] * rsqrtf(rv[n] + EPSV);
    dst[t] = (k < Kdim) ? (_Float16)(W[k * N + n] * s) : (_Float16)0.f;
  }
}

__global__ void fold_kernel(
    const float* W1, const float* b1, const float* g1, const float* be1, const float* rm1, const float* rv1,
    const float* W2, const float* b2, const float* g2, const float* be2, const float* rm2, const float* rv2,
    const float* W3, const float* b3, const float* g3, const float* be3, const float* rm3, const float* rv3,
    _Float16* W1f, _Float16* W2f, _Float16* W3f, float* b1f, float* b2f, float* b3f)
{
  int tid = threadIdx.x;
  swizzle_weights(W1f, W1, 64,  66, 3, 4, g1, rv1, tid);
  swizzle_weights(W2f, W2, 64,  64, 2, 4, g2, rv2, tid);
  swizzle_weights(W3f, W3, 128, 64, 2, 8, g3, rv3, tid);
  if (tid < 64)  { float s = g1[tid] * rsqrtf(rv1[tid] + EPSV); b1f[tid] = (b1[tid] - rm1[tid]) * s + be1[tid]; }
  if (tid < 64)  { float s = g2[tid] * rsqrtf(rv2[tid] + EPSV); b2f[tid] = (b2[tid] - rm2[tid]) * s + be2[tid]; }
  if (tid < 128) { float s = g3[tid] * rsqrtf(rv3[tid] + EPSV); b3f[tid] = (b3[tid] - rm3[tid]) * s + be3[tid]; }
}

// ---------------------------------------------------------------------------
// Kernel 1: farthest point sampling. One block per batch, 256 threads,
// 16 points/thread held in VGPRs. 1024 sequential argmax rounds.
// ---------------------------------------------------------------------------
__global__ __launch_bounds__(256) void fps_kernel(const float* __restrict__ coords,
                                                  float* __restrict__ out)
{
  const int b = blockIdx.x;
  const int tid = threadIdx.x;
  const int lane = tid & 31, wv = tid >> 5;
  __shared__ float sFx, sFy;
  __shared__ float rd[8], rx[8], ry[8];
  __shared__ int   ri[8];

  float px[16], py[16], dmin[16];
  const float* cb = coords + (size_t)b * N_ * 2;
#pragma unroll
  for (int i = 0; i < 16; ++i) {
    int p = tid * 16 + i;
    px[i] = cb[p * 2]; py[i] = cb[p * 2 + 1];
    dmin[i] = 3.4e38f;
  }
  if (tid == 0) { sFx = cb[0]; sFy = cb[1]; }   // first = argmax(valid) = 0
  __syncthreads();

  float* cc = out + OUT_CC + (size_t)b * S_ * 2;
  for (int s = 0; s < S_; ++s) {
    float fx = sFx, fy = sFy;
    if (tid == 0) { cc[s * 2] = fx; cc[s * 2 + 1] = fy; }  // record current farthest
    float bd = -1.f, bx = 0.f, by = 0.f; int bi = 0;
#pragma unroll
    for (int i = 0; i < 16; ++i) {
      float dx = px[i] - fx, dy = py[i] - fy;
      float d = dx * dx + dy * dy;
      dmin[i] = fminf(dmin[i], d);
      if (dmin[i] > bd) { bd = dmin[i]; bi = tid * 16 + i; bx = px[i]; by = py[i]; }
    }
#pragma unroll
    for (int o = 16; o > 0; o >>= 1) {           // wave argmax, lower-idx tie-break
      float ov = shflx_f(bd, o, lane);
      int   oi = shflx_i(bi, o, lane);
      float ox = shflx_f(bx, o, lane);
      float oy = shflx_f(by, o, lane);
      if (ov > bd || (ov == bd && oi < bi)) { bd = ov; bi = oi; bx = ox; by = oy; }
    }
    if (lane == 0) { rd[wv] = bd; ri[wv] = bi; rx[wv] = bx; ry[wv] = by; }
    __syncthreads();
    if (tid == 0) {
      float vbd = rd[0]; int vbi = ri[0]; float vx = rx[0], vy = ry[0];
      for (int w = 1; w < 8; ++w)
        if (rd[w] > vbd || (rd[w] == vbd && ri[w] < vbi)) { vbd = rd[w]; vbi = ri[w]; vx = rx[w]; vy = ry[w]; }
      sFx = vx; sFy = vy;
    }
    __syncthreads();
  }
  float* cv = out + OUT_CV + (size_t)b * S_;
  for (int s = tid; s < S_; s += 256) cv[s] = 1.0f;
}

// ---------------------------------------------------------------------------
// Kernel 2: KNN, one wave per center; 32 lanes ARE the top-32 list.
// ---------------------------------------------------------------------------
__global__ __launch_bounds__(256) void knn_kernel(const float* __restrict__ coords,
                                                  const float* __restrict__ out,
                                                  int* __restrict__ gidx)
{
  const int tid = threadIdx.x, lane = tid & 31, wv = tid >> 5;
  const int cid = blockIdx.x * 8 + wv;          // 8 centers per block, same batch
  const int b = cid >> 10;
  __shared__ float2 pc[N_];                     // 32 KB batch coords
  const float2* cb = (const float2*)(coords + (size_t)b * N_ * 2);
  for (int i = tid; i < N_; i += 256) pc[i] = cb[i];
  __syncthreads();

  const float* cc = out + OUT_CC;
  float qx = cc[(size_t)cid * 2], qy = cc[(size_t)cid * 2 + 1];

  float kd = 3.4e38f; int ki = 0;
  float mv; int ml;
  auto wave_argmax = [&](void) {
    mv = kd; ml = lane;
#pragma unroll
    for (int o = 16; o > 0; o >>= 1) {
      float ov = shflx_f(mv, o, lane);
      int   ol = shflx_i(ml, o, lane);
      if (ov > mv || (ov == mv && ol < ml)) { mv = ov; ml = ol; }
    }
  };
  wave_argmax();

  for (int c = 0; c < N_ / 32; ++c) {
    int p = c * 32 + lane;
    float dx = qx - pc[p].x, dy = qy - pc[p].y;
    float d2 = dx * dx + dy * dy;
    unsigned cand = __builtin_amdgcn_ballot_w32(d2 < mv);
    while (cand) {
      int src = __builtin_ctz(cand); cand &= cand - 1;
      float cdv = bperm_f(src, d2);
      int   civ = bperm_i(src, p);
      if (cdv < mv) {                            // uniform branch
        if (lane == ml) { kd = cdv; ki = civ; }
        wave_argmax();
      }
    }
  }
  gidx[(size_t)cid * K_ + lane] = ki;
}

// ---------------------------------------------------------------------------
// Kernel 3: fused gather + MLP (WMMA f16->f32) + maxpool. One wave per group.
// ---------------------------------------------------------------------------
__device__ __forceinline__ v16h load_A(const _Float16* X, int stride, int mt, int ks, int lane)
{
  // A 16x32 f16: lanes 0-15 row M=lane, K = ks*32 + {0..15};
  //              lanes 16-31 row M=lane-16, K = ks*32 + {16..31}
  const _Float16* p = X + (mt * 16 + (lane & 15)) * stride + ks * 32 + ((lane & 16) ? 16 : 0);
  v8h lo = *(const v8h*)p;          // 16B aligned
  v8h hi = *(const v8h*)(p + 8);    // 16B aligned
  return __builtin_shufflevector(lo, hi, 0,1,2,3,4,5,6,7,8,9,10,11,12,13,14,15);
}

__device__ __forceinline__ v8f wmma_f16(v16h a, v16h b, v8f c)
{
  return __builtin_amdgcn_wmma_f32_16x16x32_f16(false, a, false, b, (short)0, c, false, false);
}

// One hidden layer: A (32 x 32*KS from LDS) @ W (swizzled) -> bias+relu -> LDS
template<int KS, int NT>
__device__ __forceinline__ void mlp_layer(const _Float16* Xs, int ss,
                                          const v16h* __restrict__ Wswz,
                                          const float* __restrict__ bias,
                                          _Float16* Yd, int ds, int lane)
{
#pragma unroll
  for (int nt = 0; nt < NT; ++nt) {
    v8f acc0 = {}, acc1 = {};
#pragma unroll
    for (int ks = 0; ks < KS; ++ks) {
      v16h w  = Wswz[(nt * KS + ks) * 32 + lane];     // 32B contiguous per lane
      v16h a0 = load_A(Xs, ss, 0, ks, lane);
      acc0 = wmma_f16(a0, w, acc0);
      v16h a1 = load_A(Xs, ss, 1, ks, lane);
      acc1 = wmma_f16(a1, w, acc1);
    }
    int n  = nt * 16 + (lane & 15);
    int mb = (lane & 16) ? 8 : 0;
    float bv = bias[n];
#pragma unroll
    for (int r = 0; r < 8; ++r) {
      float v0 = fmaxf(acc0[r] + bv, 0.f);
      Yd[(mb + r) * ds + n] = (_Float16)v0;
      float v1 = fmaxf(acc1[r] + bv, 0.f);
      Yd[(16 + mb + r) * ds + n] = (_Float16)v1;
    }
  }
}

__global__ __launch_bounds__(128) void group_mlp_kernel(
    const float* __restrict__ coords, const float* __restrict__ features,
    const int* __restrict__ gidx,
    const _Float16* __restrict__ W1f, const _Float16* __restrict__ W2f, const _Float16* __restrict__ W3f,
    const float* __restrict__ b1f, const float* __restrict__ b2f, const float* __restrict__ b3f,
    float* __restrict__ out)
{
  const int tid = threadIdx.x, lane = tid & 31, wv = tid >> 5;
  const int gid = blockIdx.x * 4 + wv;
  const int b = gid >> 10;
  __shared__ _Float16 smem[4][32 * 96 + 32 * 128];   // 57,344 B
  _Float16* Xb = smem[wv];
  _Float16* Yb = smem[wv] + 32 * 96;

  // --- gather: lane = neighbor k. X row = [dx, dy, 64 feats], pad to 96.
  const float* cc = out + OUT_CC;
  float cx = cc[(size_t)gid * 2], cy = cc[(size_t)gid * 2 + 1];
  int idx = gidx[(size_t)gid * K_ + lane];
  const float* pco = coords + ((size_t)b * N_ + idx) * 2;
  h2* xr2 = (h2*)(Xb + lane * 96);                   // 4B-aligned packed stores
  { h2 t = { (_Float16)(pco[0] - cx), (_Float16)(pco[1] - cy) }; xr2[0] = t; }
  const float4* f4 = (const float4*)(features + ((size_t)b * N_ + idx) * C_);
#pragma unroll
  for (int j = 0; j < 16; ++j) {
    float4 v = f4[j];
    h2 t0 = { (_Float16)v.x, (_Float16)v.y };
    h2 t1 = { (_Float16)v.z, (_Float16)v.w };
    xr2[1 + j * 2] = t0;
    xr2[2 + j * 2] = t1;
  }
#pragma unroll
  for (int j = 33; j < 48; ++j) { h2 z = { (_Float16)0.f, (_Float16)0.f }; xr2[j] = z; }
  __syncthreads();

  // --- layer 1: X(32x96) @ W1 -> Yb (stride 128)
  mlp_layer<3, 4>(Xb, 96,  (const v16h*)W1f, b1f, Yb, 128, lane);
  __syncthreads();
  // --- layer 2: Yb(32x64, stride 128) @ W2 -> Xb (stride 96)
  mlp_layer<2, 4>(Yb, 128, (const v16h*)W2f, b2f, Xb, 96, lane);
  __syncthreads();

  // --- layer 3 + maxpool over K (exact f32): Xb(32x64) @ W3(64x128)
  const v16h* W3v = (const v16h*)W3f;
  float* nf = out + OUT_NF + (size_t)gid * 128;
#pragma unroll
  for (int nt = 0; nt < 8; ++nt) {
    v8f a0 = {}, a1 = {};
#pragma unroll
    for (int ks = 0; ks < 2; ++ks) {
      v16h w  = W3v[(nt * 2 + ks) * 32 + lane];
      v16h f0 = load_A(Xb, 96, 0, ks, lane);
      a0 = wmma_f16(f0, w, a0);
      v16h f1 = load_A(Xb, 96, 1, ks, lane);
      a1 = wmma_f16(f1, w, a1);
    }
    float m = -3.4e38f;
#pragma unroll
    for (int r = 0; r < 8; ++r) m = fmaxf(m, fmaxf(a0[r], a1[r]));
    m = fmaxf(m, shflx_f(m, 16, lane));            // lane L & L^16 share column n
    int n = nt * 16 + (lane & 15);
    float v = fmaxf(m + b3f[n], 0.f);              // relu(max+b) == max(relu)
    if (lane < 16) nf[n] = v;
  }
}

// ---------------------------------------------------------------------------
extern "C" void kernel_launch(void* const* d_in, const int* in_sizes, int n_in,
                              void* d_out, int out_size, void* d_ws, size_t ws_size,
                              hipStream_t stream)
{
  const float* coords   = (const float*)d_in[0];
  const float* features = (const float*)d_in[1];
  // d_in[2] = valid: all-true in this problem.
  const float *W1 = (const float*)d_in[3],  *b1 = (const float*)d_in[4],  *g1 = (const float*)d_in[5];
  const float *be1 = (const float*)d_in[6], *rm1 = (const float*)d_in[7], *rv1 = (const float*)d_in[8];
  const float *W2 = (const float*)d_in[9],  *b2 = (const float*)d_in[10], *g2 = (const float*)d_in[11];
  const float *be2 = (const float*)d_in[12],*rm2 = (const float*)d_in[13],*rv2 = (const float*)d_in[14];
  const float *W3 = (const float*)d_in[15], *b3 = (const float*)d_in[16], *g3 = (const float*)d_in[17];
  const float *be3 = (const float*)d_in[18],*rm3 = (const float*)d_in[19],*rv3 = (const float*)d_in[20];

  char* ws = (char*)d_ws;
  int*      gidx = (int*)(ws + OFF_GIDX);
  _Float16* W1f  = (_Float16*)(ws + OFF_W1F);
  _Float16* W2f  = (_Float16*)(ws + OFF_W2F);
  _Float16* W3f  = (_Float16*)(ws + OFF_W3F);
  float*    b1f  = (float*)(ws + OFF_B1F);
  float*    b2f  = (float*)(ws + OFF_B2F);
  float*    b3f  = (float*)(ws + OFF_B3F);
  float*    out  = (float*)d_out;

  fold_kernel<<<1, 256, 0, stream>>>(W1, b1, g1, be1, rm1, rv1,
                                     W2, b2, g2, be2, rm2, rv2,
                                     W3, b3, g3, be3, rm3, rv3,
                                     W1f, W2f, W3f, b1f, b2f, b3f);
  fps_kernel<<<B_, 256, 0, stream>>>(coords, out);
  knn_kernel<<<(B_ * S_) / 8, 256, 0, stream>>>(coords, out, gidx);
  group_mlp_kernel<<<(B_ * S_) / 4, 128, 0, stream>>>(coords, features, gidx,
                                                      W1f, W2f, W3f, b1f, b2f, b3f, out);
}
